// DTransformer_35527969473068
// MI455X (gfx1250) — compile-verified, matrix operand order checked by reading, not run on hardware
//
#include <hip/hip_runtime.h>
#include <hip/hip_bf16.h>
#include <math.h>

// ---------------------------------------------------------------------------
// DTransformer forward for MI455X (gfx1250, wave32, WMMA).
// All GEMMs run through v_wmma_f32_16x16x32_f16 (fp32 accum).
// Tiles are staged fp32 -> f16 into LDS in *fragment-major* order so each
// lane's v16h fragment is one contiguous 32-byte ds_load; LDS is
// double-buffered and global loads are software-pipelined.
// All GEMM shapes here are multiples of the 64x64x32 tile -> no bounds checks.
// ---------------------------------------------------------------------------

typedef __attribute__((ext_vector_type(16))) _Float16 v16h;
typedef __attribute__((ext_vector_type(8)))  float    v8f;

#define L_SEQ 1024
#define DMODEL 1024
#define NHEAD 16
#define DVH 64
#define DMLP 4096
#define NVOCAB 32000
#define NLAYER 4

// ---------------------------------------------------------------------------
// Generic batched GEMM:  C[b] = epilogue( A[b] (MxK) * B[b] (KxN or NxK^T) )
// epilogue: (+ bias[n]) -> (gelu?) -> (+ res[m,n]) -> (+ res2[m,n])
// Block: 128 threads = 4 waves; tile 64x64; K-step 32.
// Each wave owns a 32x32 quadrant = 2x2 WMMA 16x16 accumulators.
// REQUIRES: M%64==0, N%64==0, K%32==0, rows 16B-aligned (true for all calls).
// ---------------------------------------------------------------------------
__global__ __launch_bounds__(128) void gemm_wmma_f16(
    int M, int N, int K,
    const float* __restrict__ A, int lda, long long sA,
    const float* __restrict__ B, int ldb, long long sB, int transB,
    float* __restrict__ C, int ldc, long long sC,
    const float* __restrict__ bias, long long sBias,
    const float* __restrict__ res,  int ldres,
    const float* __restrict__ res2, int ldres2,
    int act)
{
    // fragment-major: [buf][16x16 tile][lane][16 halves]  (32B / lane, b128-able)
    __shared__ _Float16 AsF[2][4][32][16];
    __shared__ _Float16 BsF[2][4][32][16];

    const int b = blockIdx.z;
    A += (long long)b * sA;
    B += (long long)b * sB;
    C += (long long)b * sC;

    const int n0 = blockIdx.x * 64;
    const int m0 = blockIdx.y * 64;
    const int tid  = threadIdx.x;
    const int lane = tid & 31;
    const int wave = tid >> 5;
    const int wm = wave >> 1;          // wave row (0..1)
    const int wn = wave & 1;           // wave col (0..1)
    const int lg = lane >> 4;          // lane group 0/1
    const int lm = lane & 15;

    v8f acc[2][2] = {};
    float4 av[4], bv[4];

    // ---- global -> registers (vector b128 loads, no predication) ----
    auto loadA = [&](int kk) {
#pragma unroll
        for (int i = 0; i < 4; ++i) {
            int qi = tid + i * 128;            // 0..511 float4s of 64x32 tile
            int r = qi >> 3, c4 = qi & 7;
            av[i] = *(const float4*)(A + (long long)(m0 + r) * lda + kk + c4 * 4);
        }
    };
    auto loadB = [&](int kk) {
        if (transB) {
#pragma unroll
            for (int i = 0; i < 4; ++i) {
                int qi = tid + i * 128;        // n-major, contiguous along k
                int n = qi >> 3, k4 = qi & 7;
                bv[i] = *(const float4*)(B + (long long)(n0 + n) * ldb + kk + k4 * 4);
            }
        } else {
#pragma unroll
            for (int i = 0; i < 4; ++i) {
                int qi = tid + i * 128;        // k-major, contiguous along n
                int r = qi >> 4, c4 = qi & 15;
                bv[i] = *(const float4*)(B + (long long)(kk + r) * ldb + n0 + c4 * 4);
            }
        }
    };

    // k index -> (lane group, element) per documented 16-bit WMMA layout:
    //   j<8 : k = lg*8 + j       j>=8 : k = 16 + lg*8 + (j-8)
    // inverse: lg = (k>>3)&1 ;  j = (k&7) + ((k>>4)<<3)
    auto storeA = [&](int buf) {
#pragma unroll
        for (int i = 0; i < 4; ++i) {
            int qi = tid + i * 128;
            int r = qi >> 3, c4 = qi & 7;
            int mt = r >> 4, lr_ = r & 15;
            const float* f = (const float*)&av[i];
#pragma unroll
            for (int e = 0; e < 4; ++e) {
                int c = c4 * 4 + e;
                AsF[buf][mt][(((c >> 3) & 1) << 4) + lr_][(c & 7) + ((c >> 4) << 3)] =
                    (_Float16)f[e];
            }
        }
    };
    auto storeB = [&](int buf) {
        if (transB) {
#pragma unroll
            for (int i = 0; i < 4; ++i) {
                int qi = tid + i * 128;
                int n = qi >> 3, k4 = qi & 7;
                int nt = n >> 4, ln_ = n & 15;
                const float* f = (const float*)&bv[i];
#pragma unroll
                for (int e = 0; e < 4; ++e) {
                    int kx = k4 * 4 + e;
                    BsF[buf][nt][(((kx >> 3) & 1) << 4) + ln_][(kx & 7) + ((kx >> 4) << 3)] =
                        (_Float16)f[e];
                }
            }
        } else {
#pragma unroll
            for (int i = 0; i < 4; ++i) {
                int qi = tid + i * 128;
                int r = qi >> 4, c4 = qi & 15;
                int lgk = (r >> 3) & 1, jk = (r & 7) + ((r >> 4) << 3);
                const float* f = (const float*)&bv[i];
#pragma unroll
                for (int e = 0; e < 4; ++e) {
                    int n = c4 * 4 + e;
                    BsF[buf][n >> 4][(lgk << 4) + (n & 15)][jk] = (_Float16)f[e];
                }
            }
        }
    };

    // ---- prologue ----
    loadA(0); loadB(0);
    storeA(0); storeB(0);
    __syncthreads();

    int p = 0;
    for (int kk = 0; kk < K; kk += 32) {
        const bool haveNext = (kk + 32 < K);
        if (haveNext) {                        // pipeline: fetch next tile
            loadA(kk + 32);
            loadB(kk + 32);
            if (kk + 64 < K) {                 // global_prefetch_b8 for k+2 tile
                __builtin_prefetch(A + (long long)(m0 + (tid >> 1)) * lda + kk + 64 + (tid & 1) * 16, 0, 1);
                if (!transB)
                    __builtin_prefetch(B + (long long)(kk + 64 + (tid >> 2)) * ldb + n0 + (tid & 3) * 16, 0, 1);
            }
        }

        // ---- fragments: one contiguous 32B ds read per lane per fragment ----
        const v16h* Ap = (const v16h*)&AsF[p][0][0][0];
        const v16h* Bp = (const v16h*)&BsF[p][0][0][0];
        v16h a0 = Ap[(wm * 2 + 0) * 32 + lane];
        v16h a1 = Ap[(wm * 2 + 1) * 32 + lane];
        v16h b0 = Bp[(wn * 2 + 0) * 32 + lane];
        v16h b1 = Bp[(wn * 2 + 1) * 32 + lane];

        acc[0][0] = __builtin_amdgcn_wmma_f32_16x16x32_f16(false, a0, false, b0, (short)0, acc[0][0], false, false);
        acc[0][1] = __builtin_amdgcn_wmma_f32_16x16x32_f16(false, a0, false, b1, (short)0, acc[0][1], false, false);
        acc[1][0] = __builtin_amdgcn_wmma_f32_16x16x32_f16(false, a1, false, b0, (short)0, acc[1][0], false, false);
        acc[1][1] = __builtin_amdgcn_wmma_f32_16x16x32_f16(false, a1, false, b1, (short)0, acc[1][1], false, false);

        if (haveNext) { storeA(p ^ 1); storeB(p ^ 1); }
        __syncthreads();
        p ^= 1;
    }

    // ---- epilogue: C/D layout -> (m = r + lg*8, n = lm) within each 16x16 ----
#pragma unroll
    for (int mt = 0; mt < 2; ++mt)
#pragma unroll
        for (int nt = 0; nt < 2; ++nt)
#pragma unroll
            for (int r = 0; r < 8; ++r) {
                const int m = m0 + wm * 32 + mt * 16 + r + (lg << 3);
                const int n = n0 + wn * 32 + nt * 16 + lm;
                float v = acc[mt][nt][r];
                if (bias) v += bias[(long long)b * sBias + n];
                if (act == 1) {  // tanh-approx GELU (jax.nn.gelu default)
                    float u = v;
                    v = 0.5f * u * (1.f + tanhf(0.7978845608028654f *
                                                (u + 0.044715f * u * u * u)));
                }
                if (res)  v += res [(long long)m * ldres  + n];
                if (res2) v += res2[(long long)m * ldres2 + n];
                C[(long long)m * ldc + n] = v;
            }
}

// ---------------------------------------------------------------------------
// x = word_emb[ids] + pos_emb
// ---------------------------------------------------------------------------
__global__ __launch_bounds__(256) void embed_kernel(
    const int* __restrict__ ids, const float* __restrict__ we,
    const float* __restrict__ pe, float* __restrict__ x)
{
    const int row = blockIdx.x;
    const long long wb = (long long)ids[row] * DMODEL;
    const long long pb = (long long)row * DMODEL;
    for (int d = threadIdx.x; d < DMODEL; d += 256)
        x[pb + d] = we[wb + d] + pe[pb + d];
}

// ---------------------------------------------------------------------------
// LayerNorm (biased std, NO epsilon — matches reference)
// ---------------------------------------------------------------------------
__global__ __launch_bounds__(256) void ln_kernel(
    const float* __restrict__ x, const float* __restrict__ g,
    const float* __restrict__ bb, float* __restrict__ y, int Dn)
{
    __shared__ float red[256];
    const int row = blockIdx.x;
    const float* xr = x + (long long)row * Dn;
    float s = 0.f;
    for (int i = threadIdx.x; i < Dn; i += 256) s += xr[i];
    red[threadIdx.x] = s; __syncthreads();
    for (int o = 128; o > 0; o >>= 1) {
        if (threadIdx.x < o) red[threadIdx.x] += red[threadIdx.x + o];
        __syncthreads();
    }
    const float mu = red[0] / (float)Dn;
    __syncthreads();
    s = 0.f;
    for (int i = threadIdx.x; i < Dn; i += 256) { float d = xr[i] - mu; s += d * d; }
    red[threadIdx.x] = s; __syncthreads();
    for (int o = 128; o > 0; o >>= 1) {
        if (threadIdx.x < o) red[threadIdx.x] += red[threadIdx.x + o];
        __syncthreads();
    }
    const float inv = 1.f / sqrtf(red[0] / (float)Dn);
    for (int i = threadIdx.x; i < Dn; i += 256)
        y[(long long)row * Dn + i] = g[i] * (xr[i] - mu) * inv + bb[i];
}

// ---------------------------------------------------------------------------
// Causal softmax in place on scores s[h][q][k], scale applied pre-softmax.
// grid = (L, H)
// ---------------------------------------------------------------------------
__global__ __launch_bounds__(256) void softmax_causal_kernel(
    float* __restrict__ s, int L, float scale)
{
    __shared__ float red[256];
    const int q = blockIdx.x, h = blockIdx.y;
    float* row = s + ((long long)h * L + q) * L;
    const int nvalid = q + 1;

    float mx = -3.4e38f;
    for (int i = threadIdx.x; i < nvalid; i += 256) mx = fmaxf(mx, row[i] * scale);
    red[threadIdx.x] = mx; __syncthreads();
    for (int o = 128; o > 0; o >>= 1) {
        if (threadIdx.x < o) red[threadIdx.x] = fmaxf(red[threadIdx.x], red[threadIdx.x + o]);
        __syncthreads();
    }
    mx = red[0]; __syncthreads();

    float sum = 0.f;
    for (int i = threadIdx.x; i < nvalid; i += 256) {
        float e = __expf(row[i] * scale - mx);
        row[i] = e;
        sum += e;
    }
    red[threadIdx.x] = sum; __syncthreads();
    for (int o = 128; o > 0; o >>= 1) {
        if (threadIdx.x < o) red[threadIdx.x] += red[threadIdx.x + o];
        __syncthreads();
    }
    const float inv = 1.f / red[0];
    for (int i = threadIdx.x; i < nvalid; i += 256) row[i] *= inv;
    for (int i = nvalid + (int)threadIdx.x; i < L; i += 256) row[i] = 0.f;
}

// ---------------------------------------------------------------------------
// Faithful overlapping head-write: y[:, c] = o[c][:,0] for c<15,
// o[15][:, c-15] for 15<=c<79, else 0.  o is (H, L, DV).
// ---------------------------------------------------------------------------
__global__ __launch_bounds__(256) void build_y_kernel(
    const float* __restrict__ o, float* __restrict__ y, int L)
{
    const int row = blockIdx.x;
    for (int c = threadIdx.x; c < DMODEL; c += 256) {
        float v = 0.f;
        if (c < NHEAD - 1)
            v = o[((long long)c * L + row) * DVH];                    // col 0 of head c
        else if (c < NHEAD - 1 + DVH)
            v = o[((long long)(NHEAD - 1) * L + row) * DVH + (c - (NHEAD - 1))];
        y[(long long)row * DMODEL + c] = v;
    }
}

// ---------------------------------------------------------------------------
// Row softmax over vocab; logits (L x V) -> out (L x V)
// ---------------------------------------------------------------------------
__global__ __launch_bounds__(256) void softmax_rows_kernel(
    const float* __restrict__ logits, float* __restrict__ out, int Vn)
{
    __shared__ float red[256];
    const int row = blockIdx.x;
    const float* lr = logits + (long long)row * Vn;
    float* orow = out + (long long)row * Vn;

    float mx = -3.4e38f;
    for (int i = threadIdx.x; i < Vn; i += 256) mx = fmaxf(mx, lr[i]);
    red[threadIdx.x] = mx; __syncthreads();
    for (int o = 128; o > 0; o >>= 1) {
        if (threadIdx.x < o) red[threadIdx.x] = fmaxf(red[threadIdx.x], red[threadIdx.x + o]);
        __syncthreads();
    }
    mx = red[0]; __syncthreads();

    float sum = 0.f;
    for (int i = threadIdx.x; i < Vn; i += 256) sum += __expf(lr[i] - mx);
    red[threadIdx.x] = sum; __syncthreads();
    for (int o = 128; o > 0; o >>= 1) {
        if (threadIdx.x < o) red[threadIdx.x] += red[threadIdx.x + o];
        __syncthreads();
    }
    const float inv = 1.f / red[0];
    for (int i = threadIdx.x; i < Vn; i += 256)
        orow[i] = __expf(lr[i] - mx) * inv;
}

// ---------------------------------------------------------------------------
// Host-side orchestration
// ---------------------------------------------------------------------------
extern "C" void kernel_launch(void* const* d_in, const int* in_sizes, int n_in,
                              void* d_out, int out_size, void* d_ws, size_t ws_size,
                              hipStream_t stream)
{
    (void)in_sizes; (void)n_in; (void)out_size; (void)ws_size;

    const int* x_ids      = (const int*)  d_in[0];
    const float* word_emb = (const float*)d_in[1];
    const float* pos_emb  = (const float*)d_in[2];
    const float* Wq = (const float*)d_in[3];
    const float* bq = (const float*)d_in[4];
    const float* Wk = (const float*)d_in[5];
    const float* bk = (const float*)d_in[6];
    const float* Wv = (const float*)d_in[7];
    const float* bv = (const float*)d_in[8];
    const float* Wo = (const float*)d_in[9];
    const float* bo = (const float*)d_in[10];
    const float* g1 = (const float*)d_in[11];
    const float* b1 = (const float*)d_in[12];
    const float* g2 = (const float*)d_in[13];
    const float* b2 = (const float*)d_in[14];
    const float* W1 = (const float*)d_in[15];
    const float* bm1 = (const float*)d_in[16];
    const float* W2 = (const float*)d_in[17];
    const float* bm2 = (const float*)d_in[18];
    const float* gf = (const float*)d_in[19];
    const float* bf = (const float*)d_in[20];
    const float* Wu = (const float*)d_in[21];
    const float* bu = (const float*)d_in[22];
    float* out = (float*)d_out;

    const int L = L_SEQ, D = DMODEL, H = NHEAD, DV = DVH, DM = DMLP, V = NVOCAB;

    // workspace layout (floats). q,k adjacent so logits can alias them.
    float* ws = (float*)d_ws;
    size_t off = 0;
    float* q    = ws + off; off += (size_t)H * L * D;   // 16.78M
    float* k    = ws + off; off += (size_t)H * L * D;   // 16.78M
    float* s    = ws + off; off += (size_t)H * L * L;   // 16.78M
    float* v    = ws + off; off += (size_t)H * L * DV;  // 1.05M
    float* o    = ws + off; off += (size_t)H * L * DV;
    float* y    = ws + off; off += (size_t)L * D;
    float* xn   = ws + off; off += (size_t)L * D;
    float* xA   = ws + off; off += (size_t)L * D;
    float* xB   = ws + off; off += (size_t)L * D;
    float* hbuf = ws + off; off += (size_t)L * DM;      // 4.19M
    float* logits = q;  // q/k dead after layers; 2*16.78M > L*V = 32.77M

    auto gemm = [&](int M_, int N_, int K_,
                    const float* A_, int lda, long long sA,
                    const float* B_, int ldb, long long sB, int tB,
                    float* C_, int ldc, long long sC,
                    const float* bias_, long long sBias,
                    const float* res_, int ldres,
                    const float* res2_, int ldres2,
                    int act, int batch) {
        dim3 grid((N_ + 63) / 64, (M_ + 63) / 64, batch);
        gemm_wmma_f16<<<grid, dim3(128), 0, stream>>>(
            M_, N_, K_, A_, lda, sA, B_, ldb, sB, tB, C_, ldc, sC,
            bias_, sBias, res_, ldres, res2_, ldres2, act);
    };

    // ---- embed ----
    embed_kernel<<<L, 256, 0, stream>>>(x_ids, word_emb, pos_emb, xA);

    const float scale = 1.0f / sqrtf((float)D);

    for (int l = 0; l < NLAYER; ++l) {
        const float* Wq_l = Wq + (size_t)l * H * D * D;
        const float* bq_l = bq + (size_t)l * H * D;
        const float* Wk_l = Wk + (size_t)l * H * D * D;
        const float* bk_l = bk + (size_t)l * H * D;
        const float* Wv_l = Wv + (size_t)l * H * D * DV;
        const float* bv_l = bv + (size_t)l * H * DV;
        const float* Wo_l = Wo + (size_t)l * D * D;     // H*DV == D
        const float* bo_l = bo + (size_t)l * D;
        const float* W1_l = W1 + (size_t)l * D * DM;
        const float* bm1_l = bm1 + (size_t)l * DM;
        const float* W2_l = W2 + (size_t)l * DM * D;
        const float* bm2_l = bm2 + (size_t)l * D;

        // LN1
        ln_kernel<<<L, 256, 0, stream>>>(xA, g1 + (size_t)l * D, b1 + (size_t)l * D, xn, D);

        // Q,K,V projections (batched over heads via grid.z)
        gemm(L, D, D, xn, D, 0, Wq_l, D, (long long)D * D, 0,
             q, D, (long long)L * D, bq_l, D, nullptr, 0, nullptr, 0, 0, H);
        gemm(L, D, D, xn, D, 0, Wk_l, D, (long long)D * D, 0,
             k, D, (long long)L * D, bk_l, D, nullptr, 0, nullptr, 0, 0, H);
        gemm(L, DV, D, xn, D, 0, Wv_l, DV, (long long)D * DV, 0,
             v, DV, (long long)L * DV, bv_l, DV, nullptr, 0, nullptr, 0, 0, H);

        // scores S = Q K^T  (transB)
        gemm(L, L, D, q, D, (long long)L * D, k, D, (long long)L * D, 1,
             s, L, (long long)L * L, nullptr, 0, nullptr, 0, nullptr, 0, 0, H);

        // causal softmax with scale
        softmax_causal_kernel<<<dim3(L, H), 256, 0, stream>>>(s, L, scale);

        // O = P V
        gemm(L, DV, L, s, L, (long long)L * L, v, DV, (long long)L * DV, 0,
             o, DV, (long long)L * DV, nullptr, 0, nullptr, 0, nullptr, 0, 0, H);

        // overlapping head writes -> y (L x D)
        build_y_kernel<<<L, 256, 0, stream>>>(o, y, L);

        // x = x + y @ Wo + bo   (xA -> xB)
        gemm(L, D, D, y, D, 0, Wo_l, D, 0, 0,
             xB, D, 0, bo_l, 0, xA, D, nullptr, 0, 0, 1);

        // LN2
        ln_kernel<<<L, 256, 0, stream>>>(xB, g2 + (size_t)l * D, b2 + (size_t)l * D, xn, D);

        // h = gelu(xn2 @ W1 + bm1)
        gemm(L, DM, D, xn, D, 0, W1_l, DM, 0, 0,
             hbuf, DM, 0, bm1_l, 0, nullptr, 0, nullptr, 0, 1, 1);

        // x = x + xn2 + h @ W2 + bm2   (xB -> xA)
        gemm(L, D, DM, hbuf, DM, 0, W2_l, D, 0, 0,
             xA, D, 0, bm2_l, 0, xB, D, xn, D, 0, 1);
    }

    // final LN
    ln_kernel<<<L, 256, 0, stream>>>(xA, gf, bf, xn, D);

    // logits = xn @ Wu + bu    (N = 32000 = 500 x 64 tiles)
    gemm(L, V, D, xn, D, 0, Wu, V, 0, 0,
         logits, V, 0, bu, 0, nullptr, 0, nullptr, 0, 0, 1);

    // softmax over vocab -> out
    softmax_rows_kernel<<<L, 256, 0, stream>>>(logits, out, V);
}
